// Attention_pooling_22874995818771
// MI455X (gfx1250) — matrix operand order repaired and checked by reading, compiled, not run on hardware
//
#include <hip/hip_runtime.h>
#include <math.h>

typedef __attribute__((ext_vector_type(16))) _Float16 v16h;
typedef __attribute__((ext_vector_type(8)))  float    v8f;

#define HIDDEN    256
#define NEG_SLOPE 0.2f
#define NTHREADS  256
#define NWAVES    (NTHREADS / 32)
#define MAXL      4096   // LDS logit capacity per segment (16 KB); spill path covers the rest

__device__ __forceinline__ int lower_bound(const int* __restrict__ b, int n, int key) {
    int lo = 0, hi = n;
    while (lo < hi) { int mid = (lo + hi) >> 1; if (b[mid] < key) lo = mid + 1; else hi = mid; }
    return lo;
}

// ---------------------------------------------------------------------------
// One workgroup owns one (contiguous) segment end-to-end:
//   pass 1: logits = leaky_relu(feat . a) via v_wmma_f32_16x16x32_f16 -> LDS
//   pass 2: block softmax stats; LDS logits overwritten with exp(l - max)
//   pass 3: weighted mean, re-reading segment rows (L2-hot) coalesced by channel
// ---------------------------------------------------------------------------
__global__ void __launch_bounds__(NTHREADS)
k_fused_attn_pool(const float* __restrict__ feat,
                  const float* __restrict__ a,
                  const int*   __restrict__ batch,
                  float*       __restrict__ out,
                  float*       __restrict__ wslog,   // global spill for logits (rarely used)
                  int n_nodes) {
    const int s    = blockIdx.x;
    const int tid  = threadIdx.x;
    const int lane = tid & 31;
    const int wid  = tid >> 5;

    __shared__ int      srange[2];
    __shared__ float    sred[NWAVES];
    __shared__ float    sbc[2];          // [0]=max, [1]=scale
    __shared__ _Float16 sa[HIDDEN];
    __shared__ float    slog[MAXL];

    if (tid == 0) {
        srange[0] = lower_bound(batch, n_nodes, s);
        srange[1] = lower_bound(batch, n_nodes, s + 1);
    }
    sa[tid] = (_Float16)a[tid];          // NTHREADS == HIDDEN
    __syncthreads();
    const int start = srange[0], end = srange[1];
    const int cnt   = end - start;

    // ---------------- pass 1: logits via WMMA ----------------
    const int ntiles = (cnt + 15) >> 4;            // wave-uniform
    const int koff   = (lane < 16) ? 0 : 8;        // A: K sub-offset per lane group
    const int bko    = (lane < 16) ? 0 : 16;       // unused; B is loaded with koff, same k order as A

    for (int t = wid; t < ntiles; t += NWAVES) {   // uniform per wave
        const int r0 = start + t * 16;
        int row = r0 + (lane & 15);                // A layout: M = lane&15
        if (row >= end) row = end - 1;             // clamp; store is guarded
        const float* frow = feat + (size_t)row * HIDDEN;

        v8f c = {};
        #pragma unroll
        for (int kc = 0; kc < HIDDEN; kc += 32) {
            v16h af, bf;
            #pragma unroll
            for (int j = 0; j < 8; ++j) {
                af[j]     = (_Float16)frow[kc + koff + j];
                af[j + 8] = (_Float16)frow[kc + 16 + koff + j];
            }
            #pragma unroll
            for (int j = 0; j < 8; ++j) {                           // broadcast column
                bf[j]     = sa[kc + koff + j];
                bf[j + 8] = sa[kc + 16 + koff + j];
            }
            c = __builtin_amdgcn_wmma_f32_16x16x32_f16(
                    false, af, false, bf, (short)0, c, false, false);
        }
        // D: VGPR j -> (M=j, lanes 0-15) / (M=8+j, lanes 16-31); all columns identical
        float v = 0.0f; int m = -1;
        #pragma unroll
        for (int j = 0; j < 8; ++j)
            if ((lane & 15) == j) { v = c[j]; m = (lane < 16) ? j : (8 + j); }
        if (m >= 0) {
            int rm = r0 + m;
            if (rm < end) {
                float L = (v > 0.0f) ? v : NEG_SLOPE * v;
                int idx = rm - start;
                if (idx < MAXL) slog[idx] = L; else wslog[rm] = L;
            }
        }
    }
    __syncthreads();

    // ---------------- pass 2: softmax stats ----------------
    float mx = -INFINITY;
    for (int i = tid; i < cnt; i += NTHREADS) {
        float L = (i < MAXL) ? slog[i] : wslog[start + i];
        mx = fmaxf(mx, L);
    }
    #pragma unroll
    for (int o = 16; o > 0; o >>= 1) mx = fmaxf(mx, __shfl_xor(mx, o, 32));
    if (lane == 0) sred[wid] = mx;
    __syncthreads();
    if (wid == 0) {
        float m2 = (lane < NWAVES) ? sred[lane] : -INFINITY;
        #pragma unroll
        for (int o = 16; o > 0; o >>= 1) m2 = fmaxf(m2, __shfl_xor(m2, o, 32));
        if (lane == 0) sbc[0] = m2;
    }
    __syncthreads();
    mx = sbc[0];
    __syncthreads();

    float sum = 0.0f;
    for (int i = tid; i < cnt; i += NTHREADS) {    // overwrite slog with exp weights
        float L = (i < MAXL) ? slog[i] : wslog[start + i];
        float e = __expf(L - mx);
        if (i < MAXL) slog[i] = e;
        sum += e;
    }
    #pragma unroll
    for (int o = 16; o > 0; o >>= 1) sum += __shfl_xor(sum, o, 32);
    if (lane == 0) sred[wid] = sum;
    __syncthreads();
    if (tid == 0) {
        float tot = 0.0f;
        #pragma unroll
        for (int w = 0; w < NWAVES; ++w) tot += sred[w];
        sbc[1] = (cnt > 0) ? 1.0f / (tot * (float)cnt) : 0.0f;
    }
    __syncthreads();
    const float scale = sbc[1];

    // ---------------- pass 3: weighted mean (rows L2-hot) ----------------
    const int h = tid;                              // hidden channel
    const float* fcol = feat + (size_t)start * HIDDEN + h;
    float acc = 0.0f;

    const int cl = (cnt < MAXL) ? cnt : MAXL;
    int i = 0;
    for (; i + 4 <= cl; i += 4) {                   // 4-way MLP, weights LDS-broadcast
        float w0 = slog[i + 0], w1 = slog[i + 1], w2 = slog[i + 2], w3 = slog[i + 3];
        acc = fmaf(w0, fcol[(size_t)(i + 0) * HIDDEN], acc);
        acc = fmaf(w1, fcol[(size_t)(i + 1) * HIDDEN], acc);
        acc = fmaf(w2, fcol[(size_t)(i + 2) * HIDDEN], acc);
        acc = fmaf(w3, fcol[(size_t)(i + 3) * HIDDEN], acc);
    }
    for (; i < cl; ++i)
        acc = fmaf(slog[i], fcol[(size_t)i * HIDDEN], acc);
    for (; i < cnt; ++i) {                          // global spill path (cnt > MAXL only)
        float w = __expf(wslog[start + i] - mx);
        acc = fmaf(w, fcol[(size_t)i * HIDDEN], acc);
    }

    out[(size_t)s * HIDDEN + h] = acc * scale;
}

// ---------------------------------------------------------------------------
extern "C" void kernel_launch(void* const* d_in, const int* in_sizes, int n_in,
                              void* d_out, int out_size, void* d_ws, size_t ws_size,
                              hipStream_t stream) {
    (void)n_in; (void)ws_size;
    const float* feat  = (const float*)d_in[0];   // [N, 256] f32
    const float* a     = (const float*)d_in[1];   // [256, 1] f32
    const int*   batch = (const int*)d_in[2];     // [N] i32 (sorted)
    float*       out   = (float*)d_out;           // [S, 256] f32

    const int n_nodes = in_sizes[2];
    const int num_seg = out_size / HIDDEN;
    float* wslog = (float*)d_ws;                  // spill buffer, n_nodes floats

    k_fused_attn_pool<<<num_seg, NTHREADS, 0, stream>>>(feat, a, batch, out, wslog, n_nodes);
}
